// HCAM_34806414967026
// MI455X (gfx1250) — matrix operand order repaired and checked by reading, not compiled, not run on hardware
//
#include <hip/hip_runtime.h>
#include <hip/hip_bf16.h>
#include <math.h>

#define LL   4096
#define BATCH 8
#define CDIM 256
#define C3   768

typedef __attribute__((ext_vector_type(16))) __bf16 v16bf;
typedef __attribute__((ext_vector_type(8)))  float  v8f;
typedef __attribute__((ext_vector_type(4)))  unsigned int v4u;

union ABf { v16bf v; v4u q[2]; unsigned int u[8]; };

__device__ __forceinline__ unsigned short f2bf(float f) {
  unsigned int u = __float_as_uint(f);
  u += 0x7FFFu + ((u >> 16) & 1u);
  return (unsigned short)(u >> 16);
}
__device__ __forceinline__ float bf2f(unsigned short h) {
  return __uint_as_float(((unsigned int)h) << 16);
}

// group tables: sizes [32,64,64,96], starts [0,32,96,160]
__device__ const int g_rowgrp[16] = {0,0,1,1,1,1,2,2,2,2,3,3,3,3,3,3};
__device__ const int g_rowm0[16]  = {0,16,0,16,32,48,0,16,32,48,0,16,32,48,64,80};
__device__ const int g_gsize[4]   = {32,64,64,96};
__device__ const int g_gstart[4]  = {0,32,96,160};
__device__ const int g_logoff[4]  = {0, 8192, 40960, 73728}; // per-batch stride g*g

// ---------------------------------------------------------------- converts
__global__ void k_f32_to_bf16(const float* __restrict__ src,
                              unsigned short* __restrict__ dst, int n) {
  int i = blockIdx.x * blockDim.x + threadIdx.x;
  if (i < n) dst[i] = f2bf(src[i]);
}

// x (b, C, L) fp32 -> (b, L, C) bf16
__global__ void k_x_to_nlc(const float* __restrict__ x,
                           unsigned short* __restrict__ xb) {
  int t = blockIdx.x * blockDim.x + threadIdx.x;   // BATCH*LL*CDIM = 2^23
  int c = t & (CDIM - 1);
  int l = (t >> 8) & (LL - 1);
  int b = t >> 20;
  xb[t] = f2bf(x[((size_t)b * CDIM + c) * LL + l]);
}

// ------------------------------------------------------------- WMMA GEMM
// O[b,l,o] = sum_c W[o,c] * Act[b,l,c]   (A=W row-major, B=Act (b,L,CI) bf16)
enum { EP_PLAIN = 0, EP_GATE = 1, EP_BIAS = 2, EP_UP = 3, EP_PROJ = 4 };

template <int MODE>
__device__ __forceinline__ void gemm_epilogue(v8f acc, int b, int l, int m0, int h,
                                              int CO, const float* bias,
                                              const unsigned short* zbf,
                                              const float* addf,
                                              unsigned short* obf, float* of) {
  size_t ob = ((size_t)b * LL + l) * CO + m0 + 8 * h;
#pragma unroll
  for (int i = 0; i < 8; ++i) {
    int o = m0 + 8 * h + i;
    float v = acc[i];
    if (MODE == EP_PLAIN) {
      obf[ob + i] = f2bf(v);
    } else if (MODE == EP_GATE) {
      float t = v + bias[o];
      float g = 0.5f * t * (1.0f + erff(t * 0.70710678f)); // exact gelu
      obf[ob + i] = f2bf(g * bf2f(zbf[ob + i]));
    } else if (MODE == EP_BIAS) {
      obf[ob + i] = f2bf(v + bias[o]);
    } else if (MODE == EP_UP) {
      obf[ob + i] = f2bf(v + bias[o] + addf[ob + i]);
    } else { // EP_PROJ: write final fp32 (b, CO, L)
      of[((size_t)b * CO + o) * LL + l] = v;
    }
  }
}

// Block: 256 threads (8 waves). All waves share weight rows m0..m0+15 which are
// staged once into LDS. Each wave computes a 16(M) x 64(N) output slab: four
// independent WMMAs per k-step (XDL back-to-back, no D->A/B hazards).
template <int MODE>
__global__ void k_gemm(const unsigned short* __restrict__ W,
                       const unsigned short* __restrict__ Act,
                       int CO, int CI,
                       const float* __restrict__ bias,
                       const unsigned short* zbf,        // (b,L,CO) for GATE
                       const float* __restrict__ addf,   // (b,L,CO) for UP
                       unsigned short* __restrict__ obf, // (b,L,CO)
                       float* __restrict__ of) {         // (b,CO,L) for PROJ
  __shared__ unsigned short As[16 * 256];                // up to 8KB
  int wave = threadIdx.x >> 5;
  int lane = threadIdx.x & 31;
  int h = lane >> 4, nn = lane & 15;
  int m0 = blockIdx.y * 16;
  int b  = blockIdx.z;
  { // stage A panel (16 x CI) into LDS cooperatively, 16B per transfer
    int rw = CI >> 3;                 // uint4s per row
    int tot = 16 * rw;
    for (int idx = threadIdx.x; idx < tot; idx += 256) {
      int r = idx / rw, cc = idx - r * rw;
      ((v4u*)As)[idx] = *(const v4u*)(W + (size_t)(m0 + r) * CI + cc * 8);
    }
  }
  __syncthreads();
  int n0 = blockIdx.x * 512 + wave * 64;
  const unsigned short* B0 = Act + ((size_t)b * LL + n0 + nn) * CI;
  const unsigned short* B1 = B0 + (size_t)16 * CI;
  const unsigned short* B2 = B0 + (size_t)32 * CI;
  const unsigned short* B3 = B0 + (size_t)48 * CI;
  __builtin_prefetch(B0, 0, 0);
  __builtin_prefetch(B1, 0, 0);
  __builtin_prefetch(B2, 0, 0);
  __builtin_prefetch(B3, 0, 0);
  const unsigned short* Alds = As + nn * CI;
  v8f acc0 = {}, acc1 = {}, acc2 = {}, acc3 = {};
  for (int k0 = 0; k0 < CI; k0 += 32) {
    ABf a;
    // A 16x32 bf16 fragment from LDS: VGPR i: K=2i+16h
    a.q[0] = *(const v4u*)(Alds + k0 + 16 * h);
    a.q[1] = *(const v4u*)(Alds + k0 + 16 * h + 8);
    // B 32x16 bf16 fragments: lane(h,n): VGPR i: K=2i+16h
    ABf f0, f1, f2, f3;
    f0.q[0] = *(const v4u*)(B0 + k0 + 16 * h);
    f0.q[1] = *(const v4u*)(B0 + k0 + 16 * h + 8);
    f1.q[0] = *(const v4u*)(B1 + k0 + 16 * h);
    f1.q[1] = *(const v4u*)(B1 + k0 + 16 * h + 8);
    f2.q[0] = *(const v4u*)(B2 + k0 + 16 * h);
    f2.q[1] = *(const v4u*)(B2 + k0 + 16 * h + 8);
    f3.q[0] = *(const v4u*)(B3 + k0 + 16 * h);
    f3.q[1] = *(const v4u*)(B3 + k0 + 16 * h + 8);
    acc0 = __builtin_amdgcn_wmma_f32_16x16x32_bf16(false, a.v, false, f0.v,
                                                   (short)0, acc0, false, false);
    acc1 = __builtin_amdgcn_wmma_f32_16x16x32_bf16(false, a.v, false, f1.v,
                                                   (short)0, acc1, false, false);
    acc2 = __builtin_amdgcn_wmma_f32_16x16x32_bf16(false, a.v, false, f2.v,
                                                   (short)0, acc2, false, false);
    acc3 = __builtin_amdgcn_wmma_f32_16x16x32_bf16(false, a.v, false, f3.v,
                                                   (short)0, acc3, false, false);
  }
  gemm_epilogue<MODE>(acc0, b, n0 + nn,      m0, h, CO, bias, zbf, addf, obf, of);
  gemm_epilogue<MODE>(acc1, b, n0 + 16 + nn, m0, h, CO, bias, zbf, addf, obf, of);
  gemm_epilogue<MODE>(acc2, b, n0 + 32 + nn, m0, h, CO, bias, zbf, addf, obf, of);
  gemm_epilogue<MODE>(acc3, b, n0 + 48 + nn, m0, h, CO, bias, zbf, addf, obf, of);
}

// ------------------------------------------------------- depthwise 3x3 conv
__global__ void k_dwconv(const unsigned short* __restrict__ in,
                         const float* __restrict__ wdw,
                         unsigned short* __restrict__ out) {
  int t = blockIdx.x * blockDim.x + threadIdx.x; // BATCH*LL*C3
  int ch = t % C3;
  int l  = (t / C3) & (LL - 1);
  int b  = t / (C3 * LL);
  int y = l >> 6, x = l & 63;
  const unsigned short* base = in + (size_t)b * LL * C3;
  float acc = 0.f;
#pragma unroll
  for (int dy = 0; dy < 3; ++dy) {
    int yy = y + dy - 1;
    if (yy < 0 || yy > 63) continue;
#pragma unroll
    for (int dx = 0; dx < 3; ++dx) {
      int xx = x + dx - 1;
      if (xx < 0 || xx > 63) continue;
      acc += bf2f(base[(size_t)(yy * 64 + xx) * C3 + ch]) * wdw[ch * 9 + dy * 3 + dx];
    }
  }
  out[t] = f2bf(acc);
}

// --------------------------------------------- q channel means + argsort rank
__global__ void k_qmean(const unsigned short* __restrict__ dw,
                        float* __restrict__ cm) {
  __shared__ float s[256];
  int ch = blockIdx.x, t = threadIdx.x;
  float sum = 0.f;
  for (int i = t; i < BATCH * LL; i += 256) {
    int b = i >> 12, l = i & (LL - 1);
    sum += bf2f(dw[((size_t)b * LL + l) * C3 + ch]);
  }
  s[t] = sum; __syncthreads();
  for (int st = 128; st > 0; st >>= 1) { if (t < st) s[t] += s[t + st]; __syncthreads(); }
  if (t == 0) cm[ch] = s[0] / (float)(BATCH * LL);
}

__global__ void k_rank(const float* __restrict__ cm, int* __restrict__ order) {
  __shared__ float s[256];
  int i = threadIdx.x;
  s[i] = cm[i]; __syncthreads();
  float mi = s[i]; int r = 0;
  for (int j = 0; j < 256; ++j) {
    float mj = s[j];
    r += (mj > mi) || (mj == mi && j < i); // stable argsort of -mean
  }
  order[r] = i;
}

// -------- l2-norm + sorted gather of qn/kn/v (both layouts) + group means
__global__ void k_norm_gather(const unsigned short* __restrict__ dw,
                              const int* __restrict__ order,
                              float* __restrict__ gm,
                              unsigned short* __restrict__ qn,
                              unsigned short* __restrict__ kn,
                              unsigned short* __restrict__ qnT,
                              unsigned short* __restrict__ knT,
                              unsigned short* __restrict__ vs) {
  __shared__ float sA[256], sB[256], sC[256], sD[256];
  __shared__ float s_iq, s_ik;
  int p = blockIdx.x & 255, b = blockIdx.x >> 8;
  int t = threadIdx.x;
  int ch = order[p];
  const unsigned short* base = dw + (size_t)b * LL * C3;
  float sqq = 0, sq = 0, sqk = 0, sk = 0;
  for (int l = t; l < LL; l += 256) {
    float qv = bf2f(base[(size_t)l * C3 + ch]);
    float kv = bf2f(base[(size_t)l * C3 + 256 + ch]);
    sqq += qv * qv; sq += qv; sqk += kv * kv; sk += kv;
  }
  sA[t] = sqq; sB[t] = sq; sC[t] = sqk; sD[t] = sk; __syncthreads();
  for (int st = 128; st > 0; st >>= 1) {
    if (t < st) { sA[t] += sA[t+st]; sB[t] += sB[t+st]; sC[t] += sC[t+st]; sD[t] += sD[t+st]; }
    __syncthreads();
  }
  if (t == 0) {
    float iq = 1.0f / fmaxf(sqrtf(sA[0]), 1e-12f);
    float ik = 1.0f / fmaxf(sqrtf(sC[0]), 1e-12f);
    s_iq = iq; s_ik = ik;
    gm[b * 256 + p] = (iq * sB[0] + ik * sD[0]) * (1.0f / LL); // mean of (qn+kn)
  }
  __syncthreads();
  float iq = s_iq, ik = s_ik;
  for (int l = t; l < LL; l += 256) {
    unsigned short q16 = f2bf(bf2f(base[(size_t)l * C3 + ch]) * iq);
    unsigned short k16 = f2bf(bf2f(base[(size_t)l * C3 + 256 + ch]) * ik);
    size_t oi = ((size_t)b * LL + l) * 256 + p;       // (b, L, C)
    size_t ti = ((size_t)b * 256 + p) * LL + l;       // (b, C, L)
    qn[oi] = q16; kn[oi] = k16;
    qnT[ti] = q16; knT[ti] = k16;
    vs[oi] = base[(size_t)l * C3 + 512 + ch];
  }
}

// --------------------------- attn logits: qn·kn over L (WMMA, K=L contraction)
// qnT/knT are (b, 256-sorted, L): both operands feed from contiguous b128 loads
__global__ void k_attn_logits(const unsigned short* __restrict__ qnT,
                              const unsigned short* __restrict__ knT,
                              const float* __restrict__ temp,
                              float* __restrict__ logits) {
  int nt = blockIdx.x, r = blockIdx.y, b = blockIdx.z;
  int grp = g_rowgrp[r], g = g_gsize[grp];
  if (nt * 16 >= g) return;                 // wave-uniform
  int m0 = g_rowm0[r], n0 = nt * 16, P0 = g_gstart[grp];
  int lane = threadIdx.x & 31;
  int h = lane >> 4, nn = lane & 15;
  const unsigned short* Arow = qnT + ((size_t)b * 256 + P0 + m0 + nn) * LL;
  const unsigned short* Brow = knT + ((size_t)b * 256 + P0 + n0 + nn) * LL;
  v8f acc = {};
  for (int k0 = 0; k0 < LL; k0 += 32) {
    ABf a, bb;
    a.q[0] = *(const v4u*)(Arow + k0 + 16 * h);
    a.q[1] = *(const v4u*)(Arow + k0 + 16 * h + 8);
    bb.q[0] = *(const v4u*)(Brow + k0 + 16 * h);
    bb.q[1] = *(const v4u*)(Brow + k0 + 16 * h + 8);
    acc = __builtin_amdgcn_wmma_f32_16x16x32_bf16(false, a.v, false, bb.v,
                                                  (short)0, acc, false, false);
  }
  float tv = temp[grp];
  float* lb = logits + g_logoff[grp] + (size_t)b * g * g;
#pragma unroll
  for (int i = 0; i < 8; ++i)
    lb[(size_t)(m0 + 8 * h + i) * g + n0 + nn] = acc[i] * tv;
}

__global__ void k_softmax(const float* __restrict__ logits,
                          unsigned short* __restrict__ attn) {
  __shared__ float s[128];
  int row = blockIdx.x;            // b*256 rows
  int b = row >> 8, p = row & 255;
  int grp = (p < 32) ? 0 : (p < 96) ? 1 : (p < 160) ? 2 : 3;
  int g = g_gsize[grp], c = p - g_gstart[grp];
  const float* lb = logits + g_logoff[grp] + (size_t)b * g * g + (size_t)c * g;
  unsigned short* ab = attn + g_logoff[grp] + (size_t)b * g * g + (size_t)c * g;
  int t = threadIdx.x;
  float v = (t < g) ? lb[t] : -3.0e38f;
  s[t] = v; __syncthreads();
  for (int st = 64; st > 0; st >>= 1) { if (t < st) s[t] = fmaxf(s[t], s[t+st]); __syncthreads(); }
  float mx = s[0]; __syncthreads();
  float e = (t < g) ? __expf(v - mx) : 0.f;
  s[t] = e; __syncthreads();
  for (int st = 64; st > 0; st >>= 1) { if (t < st) s[t] += s[t+st]; __syncthreads(); }
  float inv = 1.0f / s[0];
  if (t < g) ab[t] = f2bf(e * inv);
}

// ------------------------------- outs = attn @ v  (WMMA, K = group size)
__global__ void k_attn_out(const unsigned short* __restrict__ attn,
                           const unsigned short* __restrict__ vs,
                           float* __restrict__ outs) {
  int wave = threadIdx.x >> 5, lane = threadIdx.x & 31;
  int h = lane >> 4, nn = lane & 15;
  int r = blockIdx.y, b = blockIdx.z;
  int grp = g_rowgrp[r], g = g_gsize[grp];
  int m0 = g_rowm0[r], P0 = g_gstart[grp];
  int n0 = blockIdx.x * 128 + wave * 16;
  int l = n0 + nn;
  const unsigned short* Arow = attn + g_logoff[grp] + (size_t)b * g * g + (size_t)(m0 + nn) * g;
  const unsigned short* Brow = vs + ((size_t)b * LL + l) * 256 + P0;
  v8f acc = {};
  for (int k0 = 0; k0 < g; k0 += 32) {
    ABf a, bb;
    a.q[0] = *(const v4u*)(Arow + k0 + 16 * h);
    a.q[1] = *(const v4u*)(Arow + k0 + 16 * h + 8);
    bb.q[0] = *(const v4u*)(Brow + k0 + 16 * h);
    bb.q[1] = *(const v4u*)(Brow + k0 + 16 * h + 8);
    acc = __builtin_amdgcn_wmma_f32_16x16x32_bf16(false, a.v, false, bb.v,
                                                  (short)0, acc, false, false);
  }
  float* ob = outs + ((size_t)b * LL + l) * 256 + P0 + m0 + 8 * h;
#pragma unroll
  for (int i = 0; i < 8; ++i) ob[i] = acc[i];
}

// z = (qn + kn) + outs   -> bf16
__global__ void k_make_z(const unsigned short* __restrict__ qn,
                         const unsigned short* __restrict__ kn,
                         const float* __restrict__ outs,
                         unsigned short* __restrict__ z) {
  int t = blockIdx.x * 256 + threadIdx.x;
  z[t] = f2bf(bf2f(qn[t]) + bf2f(kn[t]) + outs[t]);
}

// qv_cache = 0.9 * mean of 4 tiled group-means, broadcast over 64x64
__global__ void k_qvcache(const float* __restrict__ gm, float* __restrict__ out2) {
  int bid = blockIdx.x;
  int b = bid >> 8, ch = bid & 255;
  const float* g = gm + b * 256;
  float a = g[ch & 31] + g[32 + (ch & 63)] + g[96 + (ch & 63)];
  if (ch < 192) a += g[160 + (ch % 96)];
  float v = 0.225f * a; // 0.9 * a / 4
  float* o = out2 + ((size_t)b * 256 + ch) * LL;
  for (int l = threadIdx.x; l < LL; l += 256) o[l] = v;
}

// ------------------------------------------------------------ host launcher
// workspace layout (bytes); total ~204 MB
static constexpr size_t OFF_WQ    = 0;            // 393216
static constexpr size_t OFF_WG    = 0x80000;      // 131072
static constexpr size_t OFF_WDN   = 0xA0000;      // 65536
static constexpr size_t OFF_WU    = 0xB0000;      // 65536
static constexpr size_t OFF_WP    = 0xC0000;      // 131072
static constexpr size_t OFF_CM    = 0xE0000;      // 1024
static constexpr size_t OFF_ORD   = 0xE0800;      // 1024
static constexpr size_t OFF_GM    = 0xE1000;      // 8192
static constexpr size_t OFF_LOG   = 0xE4000;      // 589824
static constexpr size_t OFF_ATT   = 0x180000;     // 294912
static constexpr size_t OFF_QKV   = 0x400000;     // 50331648 (b,L,768) bf16
static constexpr size_t OFF_ZBF   = OFF_QKV;              // 16.8MB, reuse after dwconv
static constexpr size_t OFF_QNT   = OFF_QKV + 0x1000000;  // (b,C,L) bf16 16.8MB
static constexpr size_t OFF_KNT   = OFF_QKV + 0x2000000;  // (b,C,L) bf16 16.8MB
static constexpr size_t OFF_DW    = 0x3800000;    // 50331648 (b,L,768) bf16
static constexpr size_t OFF_GATED = OFF_DW;       // reuse after gather
static constexpr size_t OFF_MID   = OFF_DW + 0x1800000; // (b,L,128) bf16
static constexpr size_t OFF_XBF   = 0x6C00000;    // 16777216 (b,L,256) bf16
static constexpr size_t OFF_FIN   = OFF_XBF;      // reuse after qkv gemm
static constexpr size_t OFF_QN    = 0x7C00000;    // 16777216
static constexpr size_t OFF_KN    = 0x8C00000;    // 16777216
static constexpr size_t OFF_VS    = 0x9C00000;    // 16777216
static constexpr size_t OFF_OUT   = 0xAC00000;    // 33554432 fp32 (b,L,256)

extern "C" void kernel_launch(void* const* d_in, const int* in_sizes, int n_in,
                              void* d_out, int out_size, void* d_ws, size_t ws_size,
                              hipStream_t stream) {
  const float* x     = (const float*)d_in[0];
  const float* temp  = (const float*)d_in[1];
  const float* w_qkv = (const float*)d_in[2];
  const float* w_dw  = (const float*)d_in[3];
  const float* w_prj = (const float*)d_in[4];
  const float* w_gat = (const float*)d_in[5];
  const float* b_gat = (const float*)d_in[6];
  const float* w_dwn = (const float*)d_in[7];
  const float* b_dwn = (const float*)d_in[8];
  const float* w_up  = (const float*)d_in[9];
  const float* b_up  = (const float*)d_in[10];

  char* ws = (char*)d_ws;
  unsigned short* wq_bf  = (unsigned short*)(ws + OFF_WQ);
  unsigned short* wg_bf  = (unsigned short*)(ws + OFF_WG);
  unsigned short* wd_bf  = (unsigned short*)(ws + OFF_WDN);
  unsigned short* wu_bf  = (unsigned short*)(ws + OFF_WU);
  unsigned short* wp_bf  = (unsigned short*)(ws + OFF_WP);
  float*          cm     = (float*)(ws + OFF_CM);
  int*            order  = (int*)(ws + OFF_ORD);
  float*          gm     = (float*)(ws + OFF_GM);
  float*          logits = (float*)(ws + OFF_LOG);
  unsigned short* attn   = (unsigned short*)(ws + OFF_ATT);
  unsigned short* qkv_bf = (unsigned short*)(ws + OFF_QKV);
  unsigned short* z_bf   = (unsigned short*)(ws + OFF_ZBF);
  unsigned short* qnT_bf = (unsigned short*)(ws + OFF_QNT);
  unsigned short* knT_bf = (unsigned short*)(ws + OFF_KNT);
  unsigned short* dw_bf  = (unsigned short*)(ws + OFF_DW);
  unsigned short* gated  = (unsigned short*)(ws + OFF_GATED);
  unsigned short* mid_bf = (unsigned short*)(ws + OFF_MID);
  unsigned short* x_bf   = (unsigned short*)(ws + OFF_XBF);
  unsigned short* fin_bf = (unsigned short*)(ws + OFF_FIN);
  unsigned short* qn_bf  = (unsigned short*)(ws + OFF_QN);
  unsigned short* kn_bf  = (unsigned short*)(ws + OFF_KN);
  unsigned short* vs_bf  = (unsigned short*)(ws + OFF_VS);
  float*          outs_f = (float*)(ws + OFF_OUT);

  float* out_proj = (float*)d_out;
  float* out_qv   = (float*)d_out + (size_t)BATCH * CDIM * LL;

  // weight + input conversion to bf16
  k_f32_to_bf16<<<(768*256 + 255)/256, 256, 0, stream>>>(w_qkv, wq_bf, 768*256);
  k_f32_to_bf16<<<(256*256 + 255)/256, 256, 0, stream>>>(w_gat, wg_bf, 256*256);
  k_f32_to_bf16<<<(128*256 + 255)/256, 256, 0, stream>>>(w_dwn, wd_bf, 128*256);
  k_f32_to_bf16<<<(256*128 + 255)/256, 256, 0, stream>>>(w_up,  wu_bf, 256*128);
  k_f32_to_bf16<<<(256*256 + 255)/256, 256, 0, stream>>>(w_prj, wp_bf, 256*256);
  k_x_to_nlc<<<32768, 256, 0, stream>>>(x, x_bf);

  // qkv = W_qkv @ x  (WMMA)
  k_gemm<EP_PLAIN><<<dim3(8, 48, 8), 256, 0, stream>>>(
      wq_bf, x_bf, 768, 256, nullptr, nullptr, nullptr, qkv_bf, nullptr);

  // depthwise 3x3
  k_dwconv<<<(BATCH*LL*C3)/256, 256, 0, stream>>>(qkv_bf, w_dw, dw_bf);

  // channel ordering
  k_qmean<<<256, 256, 0, stream>>>(dw_bf, cm);
  k_rank<<<1, 256, 0, stream>>>(cm, order);

  // l2norm + sorted gather (qn, kn, qnT, knT, v) + group means of (qn+kn)
  k_norm_gather<<<2048, 256, 0, stream>>>(dw_bf, order, gm, qn_bf, kn_bf,
                                          qnT_bf, knT_bf, vs_bf);

  // attention (WMMA) + softmax + outs (WMMA)
  k_attn_logits<<<dim3(6, 16, 8), 32, 0, stream>>>(qnT_bf, knT_bf, temp, logits);
  k_softmax<<<2048, 128, 0, stream>>>(logits, attn);
  k_attn_out<<<dim3(32, 16, 8), 256, 0, stream>>>(attn, vs_bf, outs_f);

  // z = cache + outs
  k_make_z<<<32768, 256, 0, stream>>>(qn_bf, kn_bf, outs_f, z_bf);

  // gated = gelu(Wg z + bg) * z  (WMMA, fused epilogue)
  k_gemm<EP_GATE><<<dim3(8, 16, 8), 256, 0, stream>>>(
      wg_bf, z_bf, 256, 256, b_gat, z_bf, nullptr, gated, nullptr);
  // mid = Wd gated + bd
  k_gemm<EP_BIAS><<<dim3(8, 8, 8), 256, 0, stream>>>(
      wd_bf, gated, 128, 256, b_dwn, nullptr, nullptr, mid_bf, nullptr);
  // fin = Wu mid + bu + outs
  k_gemm<EP_UP><<<dim3(8, 16, 8), 256, 0, stream>>>(
      wu_bf, mid_bf, 256, 128, b_up, nullptr, outs_f, fin_bf, nullptr);
  // out = Wp fin  -> (b, C, L) fp32 into d_out
  k_gemm<EP_PROJ><<<dim3(8, 16, 8), 256, 0, stream>>>(
      wp_bf, fin_bf, 256, 256, nullptr, nullptr, nullptr, nullptr, out_proj);

  // qv_cache second output
  k_qvcache<<<2048, 256, 0, stream>>>(gm, out_qv);

  (void)in_sizes; (void)n_in; (void)out_size; (void)ws_size;
}
